// BMLayer_Smax_Biased_46239617909556
// MI455X (gfx1250) — compile-verified, matrix operand order checked by reading, not run on hardware
//
#include <hip/hip_runtime.h>
#include <hip/hip_fp16.h>
#include <math.h>

typedef __attribute__((ext_vector_type(16))) _Float16 v16h;
typedef __attribute__((ext_vector_type(8)))  _Float16 v8h;
typedef __attribute__((ext_vector_type(8)))  float    v8f;

#define N_IMG 8
#define C_IN  16
#define H_IN  32
#define W_IN  32
#define FH    3
#define FW    3
#define OC    64
#define OH    30
#define OW    30
#define PIX   (OH*OW)          // 900
#define MTOT  (N_IMG*PIX)      // 7200
#define KTOT  (FH*FW*C_IN)     // 144
#define KPAD  160              // 5 * 32
#define CHW   (C_IN*H_IN*W_IN) // plane stride per image
#define HW    (H_IN*W_IN)      // plane stride per channel

// ---------------- prep kernels ----------------

__global__ void prep_a(const float* __restrict__ x, _Float16* __restrict__ ah) {
  int i = blockIdx.x * blockDim.x + threadIdx.x;
  if (i < N_IMG*CHW) {
    float v = fmaxf(x[i] + 5.0f, 1e-12f);
    ah[i] = (_Float16)v;
  }
}

// wh laid out OC-major: wh[oc*KTOT + j] so K is contiguous per output channel
__global__ void prep_w(const float* __restrict__ k, const float* __restrict__ bias,
                       const float* __restrict__ delta_x,
                       _Float16* __restrict__ wh, float* __restrict__ colconst) {
  int oc = threadIdx.x;            // 64 threads, one per OC
  float dx = *delta_x;
  float ks = 0.0f;
  for (int j = 0; j < KTOT; ++j) {
    float w = k[j*OC + oc];        // k flat is [144][64]
    ks += w;
    wh[oc*KTOT + j] = (_Float16)expf(w + 5.0f);
  }
  colconst[oc] = bias[oc] - ks * dx;
}

// ---------------- main implicit-GEMM WMMA kernel ----------------
// grid: MTOT/16 = 450 blocks, 128 threads (4 wave32 waves)
// block computes out tile [16 rows of M] x [64 OC]; wave w owns OC group w*16..w*16+15

__global__ void __launch_bounds__(128)
bm_wmma(const float* __restrict__ x, const _Float16* __restrict__ ah,
        const _Float16* __restrict__ wh, const float* __restrict__ colconst,
        const float* __restrict__ delta_w, float* __restrict__ out) {
  __shared__ _Float16 As[16 * KPAD];    // A tile, row-major, K padded to 160
  __shared__ _Float16 Wt[OC * KPAD];    // W tile transposed: [oc][K]
  __shared__ float    rs[16];           // raw-x patch sums per row
  __shared__ int      abase[16];        // patch-origin index into x/ah per row
  __shared__ int      obase[16];        // n*OC*PIX + rem per row (out addressing)

  const int tid = threadIdx.x;
  const int m0  = blockIdx.x * 16;

  // --- decode the 16 output rows ONCE (all divisions happen here) ---
  if (tid < 16) {
    int m   = m0 + tid;
    int n   = m / PIX;
    int rem = m - n * PIX;
    int oy  = rem / OW;
    int ox  = rem - oy * OW;
    abase[tid] = n * CHW + oy * W_IN + ox;    // + c*HW + kh*W_IN + kw
    obase[tid] = n * (OC * PIX) + rem;        // + oc*PIX
  }
  __syncthreads();

  // --- stage A tile: 16 x 160 halves (20 iters/thread, uniform, shift/add only) ---
  for (int idx = tid; idx < 16 * KPAD; idx += 128) {
    int row = idx / KPAD;
    int j   = idx - row * KPAD;
    _Float16 v = (_Float16)0.0f;
    if (j < KTOT) {
      int tap = j >> 4;              // (kh,kw) tap index 0..8
      int c   = j & 15;
      int kh  = tap / 3;
      int kw  = tap - kh * 3;
      v = ah[abase[row] + c * HW + kh * W_IN + kw];
    }
    As[idx] = v;
  }

  // --- stage W tile (transposed): 64 x 160 halves (80 iters/thread, uniform) ---
  for (int idx = tid; idx < OC * KPAD; idx += 128) {
    int oc = idx / KPAD;
    int j  = idx - oc * KPAD;
    Wt[idx] = (j < KTOT) ? wh[oc*KTOT + j] : (_Float16)0.0f;
  }

  // --- raw-x patch sums (one thread per output row) ---
  if (tid < 16) {
    int base = abase[tid];
    float s = 0.0f;
    for (int c = 0; c < C_IN; ++c)
      for (int kh = 0; kh < FH; ++kh)
        for (int kw = 0; kw < FW; ++kw)
          s += x[base + c * HW + kh * W_IN + kw];
    rs[tid] = s * (*delta_w);
  }
  __syncthreads();

  // --- WMMA: D = A(16x160) x B(160x16) per wave, 5 accumulating steps ---
  const int lane   = tid & 31;
  const int wave   = tid >> 5;           // 0..3
  const int ncol   = lane & 15;          // row (A) / col (B) within tile
  const int kb     = (lane >> 4) << 3;   // K sub-offset: 0 (lanes 0-15) or 8 (16-31)
  const int ocbase = wave * 16;

  const _Float16* arow = &As[ncol * KPAD];
  const _Float16* brow = &Wt[(ocbase + ncol) * KPAD];

  v8f acc = {};
  #pragma unroll
  for (int k0 = 0; k0 < KPAD; k0 += 32) {
    // fragment layout (ISA 7.12.2, 16-bit 16x32): lane holds K = kb..kb+7 and 16+kb..16+kb+7
    v8h alo = *(const v8h*)(arow + k0 + kb);
    v8h ahi = *(const v8h*)(arow + k0 + kb + 16);
    v8h blo = *(const v8h*)(brow + k0 + kb);
    v8h bhi = *(const v8h*)(brow + k0 + kb + 16);
    v16h a, b;
    #pragma unroll
    for (int i = 0; i < 8; ++i) {
      a[i] = alo[i]; a[8 + i] = ahi[i];
      b[i] = blo[i]; b[8 + i] = bhi[i];
    }
    acc = __builtin_amdgcn_wmma_f32_16x16x32_f16(
        /*neg_a=*/false, a, /*neg_b=*/false, b,
        /*c_mod=*/(short)0, acc, /*reuse_a=*/false, /*reuse_b=*/false);
  }

  // --- epilogue: D[r] at row M=(lane<16 ? r : r+8), col N=lane%16; no divisions ---
  const int oc = ocbase + ncol;
  const float cc = colconst[oc];
  const int ocoff = oc * PIX;
  #pragma unroll
  for (int r = 0; r < 8; ++r) {
    int row = (lane < 16) ? r : (r + 8);
    out[obase[row] + ocoff] = acc[r] - rs[row] + cc;
  }
}

// ---------------- launch ----------------

extern "C" void kernel_launch(void* const* d_in, const int* in_sizes, int n_in,
                              void* d_out, int out_size, void* d_ws, size_t ws_size,
                              hipStream_t stream) {
  const float* x       = (const float*)d_in[0];   // (8,16,32,32)
  const float* k       = (const float*)d_in[1];   // (3,3,16,64) -> [144][64]
  const float* bias    = (const float*)d_in[2];   // (64,)
  const float* delta_x = (const float*)d_in[3];   // scalar
  const float* delta_w = (const float*)d_in[4];   // scalar
  float* out = (float*)d_out;                     // (8,64,30,30)

  // workspace layout
  char* ws = (char*)d_ws;
  _Float16* ah       = (_Float16*)(ws);                              // 131072 * 2B
  _Float16* wh       = (_Float16*)(ws + 262144);                     // 64*144 * 2B
  float*    colconst = (float*)   (ws + 262144 + 18432);             // 64 * 4B

  const int nelem = N_IMG*CHW;                     // 131072
  prep_a<<<(nelem + 255) / 256, 256, 0, stream>>>(x, ah);
  prep_w<<<1, OC, 0, stream>>>(k, bias, delta_x, wh, colconst);
  bm_wmma<<<MTOT / 16, 128, 0, stream>>>(x, ah, wh, colconst, delta_w, out);
}
